// Transformer_18511309046426
// MI455X (gfx1250) — matrix operand order repaired and checked by reading, not compiled, not run on hardware
//
#include <hip/hip_runtime.h>
#include <hip/hip_bf16.h>

typedef _Float16 h16;
typedef __attribute__((ext_vector_type(16))) _Float16 v16h;
typedef __attribute__((ext_vector_type(8)))  _Float16 v8h;
typedef __attribute__((ext_vector_type(8)))  float    v8f;

#define LDST 40   // LDS stride in halves (80B, multiple of 16B)

__device__ inline v8f v8f_zero() { v8f z = {}; return z; }

// ---------------------------------------------------------------------------
// CDNA5 async global->LDS copy (16B per lane), tracked by ASYNCcnt.
// Generic LDS pointer low 32 bits == wave-relative LDS offset (ISA flat rule).
// ---------------------------------------------------------------------------
__device__ inline void async_copy_b128(void* lds, const void* gptr)
{
    asm volatile("global_load_async_to_lds_b128 %0, %1, off"
                 :
                 : "v"((unsigned int)(uintptr_t)lds),
                   "v"((unsigned long long)(uintptr_t)gptr)
                 : "memory");
}
__device__ inline void wait_async0()
{
    asm volatile("s_wait_asynccnt 0x0" ::: "memory");
}

// ---------------------------------------------------------------------------
// Generic f16 WMMA GEMM:  Out[M,N] = A[M,K] @ B[K,N] (+bias) (+res) (gelu?)
// flags: 1=bias(f32,[N]) 2=residual(f32,[M,N]) 4=exact GELU
// outF (f32) and/or outH (f16) may be null.
// BM x BN block tile, 8 waves (2x4), wave tile (BM/2)x(BN/4), WMMA 16x16x32.
// A tile staged with async global->LDS DMA; B tile staged transposed (VALU).
// ---------------------------------------------------------------------------
template<int BM, int BN>
__global__ __launch_bounds__(256)
void gemm_f16(const h16* __restrict__ A, int lda,
              const h16* __restrict__ Bw, int ldb,
              const float* __restrict__ bias,
              const float* __restrict__ res,
              float* __restrict__ outF,
              h16* __restrict__ outH,
              int M, int N, int K, int flags)
{
    constexpr int WMT = BM / 2;       // 64
    constexpr int WNT = BN / 4;       // 32 or 16
    constexpr int MI  = WMT / 16;     // 4
    constexpr int NI  = WNT / 16;     // 2 or 1

    __shared__ h16 As[BM * LDST];
    __shared__ h16 Bt[BN * LDST];

    const int tid  = threadIdx.x;
    const int lane = tid & 31;
    const int wave = tid >> 5;
    const int wm   = wave & 1;
    const int wn   = wave >> 1;
    const int hi   = (lane >= 16) ? 1 : 0;
    const int ln   = lane & 15;

    const int bn0 = blockIdx.x * BN;
    const int bm0 = blockIdx.y * BM;

    v8f acc[MI][NI];
    #pragma unroll
    for (int i = 0; i < MI; ++i)
        #pragma unroll
        for (int j = 0; j < NI; ++j) acc[i][j] = v8f_zero();

    for (int k0 = 0; k0 < K; k0 += 32) {
        // --- stage A tile: BM x 32, row-major, via async DMA ---
        #pragma unroll
        for (int c = 0; c < (BM * 32) / (256 * 8); ++c) {
            int id  = tid + c * 256;
            int row = (id * 8) / 32;
            int col = (id * 8) % 32;
            async_copy_b128(&As[row * LDST + col],
                            A + (size_t)(bm0 + row) * lda + k0 + col);
        }
        // --- stage B tile: 32 x BN, TRANSPOSED in LDS (Bt[n][k]) ---
        #pragma unroll
        for (int c = 0; c < (32 * BN) / (256 * 8); ++c) {
            int id  = tid + c * 256;
            int row = (id * 8) / BN;
            int col = (id * 8) % BN;
            v8h val = *(const v8h*)(Bw + (size_t)(k0 + row) * ldb + bn0 + col);
            #pragma unroll
            for (int j = 0; j < 8; ++j) Bt[(col + j) * LDST + row] = val[j];
        }
        // prefetch next A tile while B transpose settles
        if (k0 + 32 < K)
            __builtin_prefetch(A + (size_t)(bm0 + (tid >> 1)) * lda + k0 + 32 + (tid & 1) * 16, 0, 1);
        wait_async0();
        __syncthreads();

        // --- fragments (per ISA 16-bit A/B layouts) ---
        v16h af[MI], bf[NI];
        #pragma unroll
        for (int i = 0; i < MI; ++i) {
            int m  = wm * WMT + i * 16 + ln;
            int kb = hi ? 8 : 0;                       // A: lo lanes K0..7/16..23, hi K8..15/24..31
            union { v16h v; v8h h2[2]; } u;
            u.h2[0] = *(const v8h*)(&As[m * LDST + kb]);
            u.h2[1] = *(const v8h*)(&As[m * LDST + 16 + kb]);
            af[i] = u.v;
        }
        #pragma unroll
        for (int j = 0; j < NI; ++j) {
            int n  = wn * WNT + j * 16 + ln;
            int kb = hi ? 8 : 0;                       // B: lo lanes K0..7/16..23, hi K8..15/24..31
            union { v16h v; v8h h2[2]; } u;
            u.h2[0] = *(const v8h*)(&Bt[n * LDST + kb]);
            u.h2[1] = *(const v8h*)(&Bt[n * LDST + kb + 16]);
            bf[j] = u.v;
        }
        #pragma unroll
        for (int i = 0; i < MI; ++i)
            #pragma unroll
            for (int j = 0; j < NI; ++j)
                acc[i][j] = __builtin_amdgcn_wmma_f32_16x16x32_f16(
                    false, af[i], false, bf[j], (short)0, acc[i][j], false, false);
        __syncthreads();
    }

    // --- epilogue (D layout: row = base + r + 8*hi, col = base + ln) ---
    #pragma unroll
    for (int i = 0; i < MI; ++i) {
        #pragma unroll
        for (int j = 0; j < NI; ++j) {
            int col  = bn0 + wn * WNT + j * 16 + ln;
            float bv = (flags & 1) ? bias[col] : 0.f;
            #pragma unroll
            for (int r = 0; r < 8; ++r) {
                int row = bm0 + wm * WMT + i * 16 + r + (hi ? 8 : 0);
                float v = acc[i][j][r] + bv;
                if (flags & 2) v += res[(size_t)row * N + col];
                if (flags & 4) v = 0.5f * v * (1.0f + erff(v * 0.70710678118f));
                if (outF) outF[(size_t)row * N + col] = v;
                if (outH) outH[(size_t)row * N + col] = (h16)v;
            }
        }
    }
}

// ---------------------------------------------------------------------------
// Flash attention: one block = (b, h, 128 queries); 8 waves x 16 queries.
// K tile async-staged to LDS; V transposed; fp32 scores + online softmax.
// ---------------------------------------------------------------------------
__global__ __launch_bounds__(256)
void flash_attn(const h16* __restrict__ Q, const h16* __restrict__ Kh,
                const h16* __restrict__ V, h16* __restrict__ O,
                int Ntok, int Dfull)
{
    const int tid  = threadIdx.x;
    const int lane = tid & 31;
    const int wave = tid >> 5;
    const int hi   = (lane >= 16) ? 1 : 0;
    const int ln   = lane & 15;
    const int hh   = blockIdx.y;
    const int b    = blockIdx.z;
    const int qb   = blockIdx.x * 128 + wave * 16;
    const int hoff = hh * 64;

    __shared__ h16 Ks[32 * 72];       // K tile row-major [key][d], stride 72
    __shared__ h16 Vt[64 * LDST];     // V tile transposed [d][key]
    __shared__ h16 Pw[8 * 16 * LDST]; // per-wave P scratch
    h16* Pme = &Pw[wave * 16 * LDST];

    // Q fragments held in registers for the whole key loop
    v16h aQ[2];
    {
        const h16* qrow = Q + ((size_t)(b * Ntok + qb + ln)) * Dfull + hoff;
        #pragma unroll
        for (int kk = 0; kk < 2; ++kk) {
            union { v16h v; v8h h2[2]; } u;
            int kb = kk * 32 + (hi ? 8 : 0);
            u.h2[0] = *(const v8h*)(qrow + kb);
            u.h2[1] = *(const v8h*)(qrow + kb + 16);
            aQ[kk] = u.v;
        }
    }

    float mrow[8], lrow[8];
    #pragma unroll
    for (int r = 0; r < 8; ++r) { mrow[r] = -1e30f; lrow[r] = 0.f; }
    v8f Od[4];
    #pragma unroll
    for (int t = 0; t < 4; ++t) Od[t] = v8f_zero();

    for (int kb0 = 0; kb0 < Ntok; kb0 += 32) {
        // cooperative K/V staging (32 keys x 64 d, 8 halves per thread)
        {
            int row = tid >> 3;
            int col = (tid & 7) * 8;
            size_t base = ((size_t)(b * Ntok + kb0 + row)) * Dfull + hoff + col;
            async_copy_b128(&Ks[row * 72 + col], Kh + base);   // K: direct DMA
            v8h vv = *(const v8h*)(V + base);                  // V: transpose in regs
            #pragma unroll
            for (int j = 0; j < 8; ++j) Vt[(col + j) * LDST + row] = vv[j];
        }
        wait_async0();
        __syncthreads();

        // S(16x32) = Q @ K^T * scale, two 16x16 key subtiles
        v8f S[2];
        #pragma unroll
        for (int s = 0; s < 2; ++s) {
            v8f a = v8f_zero();
            #pragma unroll
            for (int kk = 0; kk < 2; ++kk) {
                union { v16h v; v8h h2[2]; } u;
                int kbb = kk * 32 + (hi ? 8 : 0);
                u.h2[0] = *(const v8h*)(&Ks[(s * 16 + ln) * 72 + kbb]);
                u.h2[1] = *(const v8h*)(&Ks[(s * 16 + ln) * 72 + kbb + 16]);
                a = __builtin_amdgcn_wmma_f32_16x16x32_f16(
                        false, aQ[kk], false, u.v, (short)0, a, false, false);
            }
            S[s] = a * 0.125f;   // DH^-0.5
        }

        // online softmax (row = query, spread over 16 lanes of each half)
        float alpha[8];
        #pragma unroll
        for (int r = 0; r < 8; ++r) {
            float tmax = fmaxf(S[0][r], S[1][r]);
            #pragma unroll
            for (int mk = 8; mk >= 1; mk >>= 1)
                tmax = fmaxf(tmax, __shfl_xor(tmax, mk, 32));
            float nm = fmaxf(mrow[r], tmax);
            alpha[r] = __expf(mrow[r] - nm);
            mrow[r]  = nm;
            float p0 = __expf(S[0][r] - nm);
            float p1 = __expf(S[1][r] - nm);
            S[0][r] = p0; S[1][r] = p1;
            float ps = p0 + p1;
            #pragma unroll
            for (int mk = 8; mk >= 1; mk >>= 1)
                ps += __shfl_xor(ps, mk, 32);
            lrow[r] = lrow[r] * alpha[r] + ps;
        }

        // P (D-layout) -> LDS -> A-layout fragment; rescale O
        #pragma unroll
        for (int s = 0; s < 2; ++s)
            #pragma unroll
            for (int r = 0; r < 8; ++r)
                Pme[(r + (hi ? 8 : 0)) * LDST + s * 16 + ln] = (h16)S[s][r];
        #pragma unroll
        for (int t = 0; t < 4; ++t)
            #pragma unroll
            for (int r = 0; r < 8; ++r)
                Od[t][r] *= alpha[r];

        union { v16h v; v8h h2[2]; } up;
        {
            int kbp = hi ? 8 : 0;
            up.h2[0] = *(const v8h*)(&Pme[ln * LDST + kbp]);
            up.h2[1] = *(const v8h*)(&Pme[ln * LDST + 16 + kbp]);
        }
        #pragma unroll
        for (int t = 0; t < 4; ++t) {
            union { v16h v; v8h h2[2]; } uv;
            int kbb = hi ? 8 : 0;
            uv.h2[0] = *(const v8h*)(&Vt[(t * 16 + ln) * LDST + kbb]);
            uv.h2[1] = *(const v8h*)(&Vt[(t * 16 + ln) * LDST + kbb + 16]);
            Od[t] = __builtin_amdgcn_wmma_f32_16x16x32_f16(
                        false, up.v, false, uv.v, (short)0, Od[t], false, false);
        }
        __syncthreads();
    }

    // finalize: divide by l, write f16 o
    #pragma unroll
    for (int r = 0; r < 8; ++r) {
        float inv = 1.0f / lrow[r];
        size_t row = (size_t)(b * Ntok + qb + r + (hi ? 8 : 0));
        #pragma unroll
        for (int t = 0; t < 4; ++t)
            O[row * Dfull + hoff + t * 16 + ln] = (h16)(Od[t][r] * inv);
    }
}

// ---------------------------------------------------------------------------
// LayerNorm over last dim (f32 in, f16 out). One block per row.
// ---------------------------------------------------------------------------
__global__ __launch_bounds__(256)
void layernorm_f16out(const float* __restrict__ X, const float* __restrict__ g,
                      const float* __restrict__ bt, h16* __restrict__ out, int Dn)
{
    __shared__ float red[256];
    int row = blockIdx.x;
    const float* x = X + (size_t)row * Dn;
    float s = 0.f;
    for (int i = threadIdx.x; i < Dn; i += 256) s += x[i];
    red[threadIdx.x] = s; __syncthreads();
    for (int st = 128; st > 0; st >>= 1) {
        if (threadIdx.x < st) red[threadIdx.x] += red[threadIdx.x + st];
        __syncthreads();
    }
    float mean = red[0] / Dn;
    __syncthreads();
    float v = 0.f;
    for (int i = threadIdx.x; i < Dn; i += 256) { float d = x[i] - mean; v += d * d; }
    red[threadIdx.x] = v; __syncthreads();
    for (int st = 128; st > 0; st >>= 1) {
        if (threadIdx.x < st) red[threadIdx.x] += red[threadIdx.x + st];
        __syncthreads();
    }
    float rstd = rsqrtf(red[0] / Dn + 1e-5f);
    for (int i = threadIdx.x; i < Dn; i += 256)
        out[(size_t)row * Dn + i] = (h16)((x[i] - mean) * rstd * g[i] + bt[i]);
}

__global__ void cast_f32_f16(const float* __restrict__ in, h16* __restrict__ out, int n)
{
    for (int i = blockIdx.x * blockDim.x + threadIdx.x; i < n; i += gridDim.x * blockDim.x)
        out[i] = (h16)in[i];
}

// [B,C,N] f32 -> [B*N, C] f16
__global__ void transpose_cast(const float* __restrict__ in, h16* __restrict__ out,
                               int Bn, int Cn, int Nn)
{
    int total = Bn * Cn * Nn;
    for (int i = blockIdx.x * blockDim.x + threadIdx.x; i < total; i += gridDim.x * blockDim.x) {
        int cc = i % Cn;
        int n  = (i / Cn) % Nn;
        int b  = i / (Cn * Nn);
        out[i] = (h16)in[(size_t)b * Cn * Nn + (size_t)cc * Nn + n];
    }
}

__global__ __launch_bounds__(256)
void relu_colsum(const float* __restrict__ Y, float* __restrict__ sums, int Nn, int Cn)
{
    __shared__ float red[256];
    int idx = blockIdx.x;           // b*Cn + c
    int b = idx / Cn, cc = idx % Cn;
    float s = 0.f;
    for (int n = threadIdx.x; n < Nn; n += 256)
        s += fmaxf(Y[((size_t)b * Nn + n) * Cn + cc], 0.f);
    red[threadIdx.x] = s; __syncthreads();
    for (int st = 128; st > 0; st >>= 1) {
        if (threadIdx.x < st) red[threadIdx.x] += red[threadIdx.x + st];
        __syncthreads();
    }
    if (threadIdx.x == 0) sums[idx] = red[0];
}

// y[B*N, C] -> out[B, C, N] with relu + token normalization
__global__ void normalize_out(const float* __restrict__ Y, const float* __restrict__ sums,
                              float* __restrict__ out, int Bn, int Cn, int Nn)
{
    int total = Bn * Cn * Nn;
    for (int i = blockIdx.x * blockDim.x + threadIdx.x; i < total; i += gridDim.x * blockDim.x) {
        int n  = i % Nn;
        int cc = (i / Nn) % Cn;
        int b  = i / (Nn * Cn);
        float v = fmaxf(Y[((size_t)b * Nn + n) * Cn + cc], 0.f);
        out[i] = v / (sums[b * Cn + cc] + 1e-6f);
    }
}

// ---------------------------------------------------------------------------
extern "C" void kernel_launch(void* const* d_in, const int* in_sizes, int n_in,
                              void* d_out, int out_size, void* d_ws, size_t ws_size,
                              hipStream_t stream)
{
    constexpr int B = 2, C = 64, N = 2048, D = 512, MLP = 2048, DEPTH = 2;
    const int M = B * N;            // 4096 rows

    const float* x    = (const float*)d_in[0];
    const float* c    = (const float*)d_in[1];
    const float* Wx   = (const float*)d_in[2];
    const float* bx   = (const float*)d_in[3];
    const float* Wc   = (const float*)d_in[4];
    const float* bc   = (const float*)d_in[5];
    const float* gnc  = (const float*)d_in[6];
    const float* bnc  = (const float*)d_in[7];
    const float* Wq   = (const float*)d_in[8];
    const float* Wk   = (const float*)d_in[9];
    const float* Wv   = (const float*)d_in[10];
    const float* Wo   = (const float*)d_in[11];
    const float* bo   = (const float*)d_in[12];
    const float* W1   = (const float*)d_in[13];
    const float* b1   = (const float*)d_in[14];
    const float* W2   = (const float*)d_in[15];
    const float* b2   = (const float*)d_in[16];
    const float* g1   = (const float*)d_in[17];
    const float* bg1  = (const float*)d_in[18];
    const float* g2   = (const float*)d_in[19];
    const float* bg2  = (const float*)d_in[20];
    const float* Wout = (const float*)d_in[21];
    const float* bout = (const float*)d_in[22];
    float* out = (float*)d_out;

    // bump allocator over workspace
    char* ws = (char*)d_ws;
    size_t off = 0;
    auto alloc = [&](size_t bytes) -> void* {
        void* p = ws + off;
        off = (off + bytes + 255) & ~(size_t)255;
        return p;
    };

    h16* Wx_h  = (h16*)alloc((size_t)C * D * 2);
    h16* Wc_h  = (h16*)alloc((size_t)C * D * 2);
    h16* Wq_h  = (h16*)alloc((size_t)DEPTH * D * D * 2);
    h16* Wk_h  = (h16*)alloc((size_t)DEPTH * D * D * 2);
    h16* Wv_h  = (h16*)alloc((size_t)DEPTH * D * D * 2);
    h16* Wo_h  = (h16*)alloc((size_t)DEPTH * D * D * 2);
    h16* W1_h  = (h16*)alloc((size_t)DEPTH * D * MLP * 2);
    h16* W2_h  = (h16*)alloc((size_t)DEPTH * MLP * D * 2);
    h16* Wout_h= (h16*)alloc((size_t)D * C * 2);
    h16* A_h   = (h16*)alloc((size_t)M * C * 2);
    h16* C_hT  = (h16*)alloc((size_t)M * C * 2);
    h16* ctx_h = (h16*)alloc((size_t)M * D * 2);
    float* x_f = (float*)alloc((size_t)M * D * 4);
    float* t_f = (float*)alloc((size_t)M * D * 4);
    h16* h_h   = (h16*)alloc((size_t)M * D * 2);
    h16* q_h   = (h16*)alloc((size_t)M * D * 2);
    h16* k_h   = (h16*)alloc((size_t)M * D * 2);
    h16* v_h   = (h16*)alloc((size_t)M * D * 2);
    h16* o_h   = (h16*)alloc((size_t)M * D * 2);
    h16* hid_h = (h16*)alloc((size_t)M * MLP * 2);
    float* y_f = (float*)alloc((size_t)M * C * 4);
    float* csum= (float*)alloc((size_t)B * C * 4);
    (void)ws_size; (void)n_in; (void)in_sizes; (void)out_size;

    auto blocks = [](int n) { return dim3((n + 255) / 256); };

    // weight casts f32 -> f16
    cast_f32_f16<<<blocks(C * D), 256, 0, stream>>>(Wx, Wx_h, C * D);
    cast_f32_f16<<<blocks(C * D), 256, 0, stream>>>(Wc, Wc_h, C * D);
    cast_f32_f16<<<blocks(DEPTH * D * D), 256, 0, stream>>>(Wq, Wq_h, DEPTH * D * D);
    cast_f32_f16<<<blocks(DEPTH * D * D), 256, 0, stream>>>(Wk, Wk_h, DEPTH * D * D);
    cast_f32_f16<<<blocks(DEPTH * D * D), 256, 0, stream>>>(Wv, Wv_h, DEPTH * D * D);
    cast_f32_f16<<<blocks(DEPTH * D * D), 256, 0, stream>>>(Wo, Wo_h, DEPTH * D * D);
    cast_f32_f16<<<blocks(DEPTH * D * MLP), 256, 0, stream>>>(W1, W1_h, DEPTH * D * MLP);
    cast_f32_f16<<<blocks(DEPTH * MLP * D), 256, 0, stream>>>(W2, W2_h, DEPTH * MLP * D);
    cast_f32_f16<<<blocks(D * C), 256, 0, stream>>>(Wout, Wout_h, D * C);

    // inputs: [B,C,N] -> [B*N,C] f16
    transpose_cast<<<blocks(B * C * N), 256, 0, stream>>>(x, A_h, B, C, N);
    transpose_cast<<<blocks(B * C * N), 256, 0, stream>>>(c, C_hT, B, C, N);

    // x = xT @ Wx + bx  -> f32 residual stream
    gemm_f16<128,128><<<dim3(D / 128, M / 128), 256, 0, stream>>>(
        A_h, C, Wx_h, D, bx, nullptr, x_f, nullptr, M, D, C, 1);
    // ctx_pre = cT @ Wc + bc
    gemm_f16<128,128><<<dim3(D / 128, M / 128), 256, 0, stream>>>(
        C_hT, C, Wc_h, D, bc, nullptr, t_f, nullptr, M, D, C, 1);
    // ctx = LN(ctx_pre)
    layernorm_f16out<<<M, 256, 0, stream>>>(t_f, gnc, bnc, ctx_h, D);

    for (int i = 0; i < DEPTH; ++i) {
        const h16* wq = Wq_h + (size_t)i * D * D;
        const h16* wk = Wk_h + (size_t)i * D * D;
        const h16* wv = Wv_h + (size_t)i * D * D;
        const h16* wo = Wo_h + (size_t)i * D * D;
        const h16* w1 = W1_h + (size_t)i * D * MLP;
        const h16* w2 = W2_h + (size_t)i * MLP * D;

        // h = LN1(x)
        layernorm_f16out<<<M, 256, 0, stream>>>(x_f, g1 + i * D, bg1 + i * D, h_h, D);
        // q = h @ Wq ; k = ctx @ Wk ; v = ctx @ Wv   (no bias)
        gemm_f16<128,128><<<dim3(D / 128, M / 128), 256, 0, stream>>>(
            h_h, D, wq, D, nullptr, nullptr, nullptr, q_h, M, D, D, 0);
        gemm_f16<128,128><<<dim3(D / 128, M / 128), 256, 0, stream>>>(
            ctx_h, D, wk, D, nullptr, nullptr, nullptr, k_h, M, D, D, 0);
        gemm_f16<128,128><<<dim3(D / 128, M / 128), 256, 0, stream>>>(
            ctx_h, D, wv, D, nullptr, nullptr, nullptr, v_h, M, D, D, 0);
        // attention
        flash_attn<<<dim3(N / 128, 8, B), 256, 0, stream>>>(q_h, k_h, v_h, o_h, N, D);
        // x = x + o @ Wo + bo
        gemm_f16<128,128><<<dim3(D / 128, M / 128), 256, 0, stream>>>(
            o_h, D, wo, D, bo + i * D, x_f, x_f, nullptr, M, D, D, 1 | 2);
        // h2 = LN2(x)
        layernorm_f16out<<<M, 256, 0, stream>>>(x_f, g2 + i * D, bg2 + i * D, h_h, D);
        // hid = gelu(h2 @ W1 + b1)  (f16 out)
        gemm_f16<128,128><<<dim3(MLP / 128, M / 128), 256, 0, stream>>>(
            h_h, D, w1, MLP, b1 + i * MLP, nullptr, nullptr, hid_h, M, MLP, D, 1 | 4);
        // x = x + hid @ W2 + b2
        gemm_f16<128,128><<<dim3(D / 128, M / 128), 256, 0, stream>>>(
            hid_h, MLP, w2, D, b2 + i * D, x_f, x_f, nullptr, M, D, MLP, 1 | 2);
    }

    // y = x @ Wout + bout
    cast_f32_f16<<<blocks(M * D), 256, 0, stream>>>(x_f, h_h, M * D);
    gemm_f16<128,64><<<dim3(C / 64, M / 128), 256, 0, stream>>>(
        h_h, D, Wout_h, C, bout, nullptr, y_f, nullptr, M, C, D, 1);

    // relu + normalize over tokens, transpose to [B,C,N]
    relu_colsum<<<B * C, 256, 0, stream>>>(y_f, csum, N, C);
    normalize_out<<<blocks(B * C * N), 256, 0, stream>>>(y_f, csum, out, B, C, N);
}
